// DilatedAttention_27685359190790
// MI455X (gfx1250) — matrix operand order repaired and checked by reading, not compiled, not run on hardware
//
#include <hip/hip_runtime.h>

typedef __attribute__((ext_vector_type(16))) __bf16 v16bf;
typedef __attribute__((ext_vector_type(8)))  float  v8f;
typedef __attribute__((ext_vector_type(2)))  float  v2f;
typedef __attribute__((ext_vector_type(2)))  __bf16 v2bf;

union ABFrag { unsigned u[8]; uint4 q[2]; v16bf v; };

__device__ __forceinline__ unsigned pk_bf16(float a, float b) {
  v2f t; t.x = a; t.y = b;
  union { v2bf v; unsigned u; } o;
  o.v = __builtin_convertvector(t, v2bf);
  return o.u;
}

__device__ __forceinline__ float fast_exp2(float x) {
  return __builtin_amdgcn_exp2f(x);  // v_exp_f32 (base-2 native)
}

// ---------------------------------------------------------------------------
// Flash attention over one dilated segment group, transposed formulation:
//   S^T = K * Q^T   (row = key, col = query row)
//   O^T = V^T * P^T (row = channel, col = query row)
// Q is pre-scaled by 0.125*log2(e) so softmax runs in base-2 (bare v_exp_f32)
// with no per-score scaling. K/V tiles are register-prefetched one iteration
// ahead so HBM latency hides behind the 16 WMMAs per 64-key step.
// grid = (16 q-tiles, 4 heads, 2*nseg instances), block = 256 (8 waves).
// ---------------------------------------------------------------------------
__global__ __launch_bounds__(256) void dilfa_flash_27685(
    const float* __restrict__ qp, const float* __restrict__ kp,
    const float* __restrict__ vp, float* __restrict__ og,
    float* __restrict__ sums,
    int s, int r, int off, int hbase, int nseg)
{
  constexpr int H = 12, D = 64, N = 8192, G = 4, SD = 2048;
  constexpr int KP = 36;  // K LDS pitch in dwords (72 halves)
  constexpr int VP = 36;  // V^T LDS pitch in dwords (72 halves)
  constexpr int NT = SD / 64;
  const int rowStride = r * H * D;

  const int qt   = blockIdx.x;
  const int hg   = blockIdx.y;
  const int z    = blockIdx.z;
  const int bb   = z / nseg;
  const int seg  = z - bb * nseg;
  const int head = hbase + hg;

  const int lane = threadIdx.x & 31;
  const int wave = threadIdx.x >> 5;
  const int m    = lane & 15;
  const int hi   = lane >> 4;

  const long base0 = ((long)(bb * N + seg * s + off) * H + head) * D;
  const float* qb = qp + base0;
  const float* kb = kp + base0;
  const float* vb = vp + base0;

  __shared__ __align__(16) unsigned short Kl[64 * 72];  // [key][d]   row-major
  __shared__ __align__(16) unsigned short Vl[64 * 72];  // [ch][key]  transposed

  const int qrow0 = qt * 128 + wave * 16;

  // Q B-frags, pre-scaled by (1/sqrt(64)) * log2(e)
  const float QS = 0.125f * 1.44269504088896f;
  ABFrag qf[2];
#pragma unroll
  for (int f = 0; f < 2; ++f) {
    const float* qr = qb + (long)(qrow0 + m) * rowStride + f * 32 + 16 * hi;
    const float4 a = *(const float4*)(qr + 0);
    const float4 b = *(const float4*)(qr + 4);
    const float4 c = *(const float4*)(qr + 8);
    const float4 d = *(const float4*)(qr + 12);
    qf[f].u[0] = pk_bf16(a.x * QS, a.y * QS); qf[f].u[1] = pk_bf16(a.z * QS, a.w * QS);
    qf[f].u[2] = pk_bf16(b.x * QS, b.y * QS); qf[f].u[3] = pk_bf16(b.z * QS, b.w * QS);
    qf[f].u[4] = pk_bf16(c.x * QS, c.y * QS); qf[f].u[5] = pk_bf16(c.z * QS, c.w * QS);
    qf[f].u[6] = pk_bf16(d.x * QS, d.y * QS); qf[f].u[7] = pk_bf16(d.z * QS, d.w * QS);
  }

  v8f o[4];
#pragma unroll
  for (int cb = 0; cb < 4; ++cb)
#pragma unroll
    for (int j = 0; j < 8; ++j) o[cb][j] = 0.0f;

  float rowmax = -1.0e30f;
  float rowsum = 0.0f;

  const unsigned* Kd = (const unsigned*)Kl;
  const unsigned* Vd = (const unsigned*)Vl;

  // register prefetch pipeline for the cooperative K/V tile loads
  const int lrow = threadIdx.x >> 2;
  const int lc0  = (threadIdx.x & 3) * 16;
  const float* kl0 = kb + (long)lrow * rowStride + lc0;
  const float* vl0 = vb + (long)lrow * rowStride + lc0;
  const long tileStride = (long)64 * rowStride;

  float4 kr0, kr1, kr2, kr3, vr0, vr1, vr2, vr3;
  {
    kr0 = *(const float4*)(kl0 + 0);  kr1 = *(const float4*)(kl0 + 4);
    kr2 = *(const float4*)(kl0 + 8);  kr3 = *(const float4*)(kl0 + 12);
    vr0 = *(const float4*)(vl0 + 0);  vr1 = *(const float4*)(vl0 + 4);
    vr2 = *(const float4*)(vl0 + 8);  vr3 = *(const float4*)(vl0 + 12);
  }

  for (int kt = 0; kt < NT; ++kt) {
    __syncthreads();
    {  // convert prefetched tile to bf16 and stage into LDS
      uint4* kd = (uint4*)((unsigned*)Kl + lrow * KP + (lc0 >> 1));
      kd[0] = make_uint4(pk_bf16(kr0.x, kr0.y), pk_bf16(kr0.z, kr0.w),
                         pk_bf16(kr1.x, kr1.y), pk_bf16(kr1.z, kr1.w));
      kd[1] = make_uint4(pk_bf16(kr2.x, kr2.y), pk_bf16(kr2.z, kr2.w),
                         pk_bf16(kr3.x, kr3.y), pk_bf16(kr3.z, kr3.w));
      unsigned short* vl = Vl + lrow;  // column = key index
      unsigned pv;
      pv = pk_bf16(vr0.x, vr0.y);
      vl[(lc0 + 0) * 72] = (unsigned short)pv;  vl[(lc0 + 1) * 72] = (unsigned short)(pv >> 16);
      pv = pk_bf16(vr0.z, vr0.w);
      vl[(lc0 + 2) * 72] = (unsigned short)pv;  vl[(lc0 + 3) * 72] = (unsigned short)(pv >> 16);
      pv = pk_bf16(vr1.x, vr1.y);
      vl[(lc0 + 4) * 72] = (unsigned short)pv;  vl[(lc0 + 5) * 72] = (unsigned short)(pv >> 16);
      pv = pk_bf16(vr1.z, vr1.w);
      vl[(lc0 + 6) * 72] = (unsigned short)pv;  vl[(lc0 + 7) * 72] = (unsigned short)(pv >> 16);
      pv = pk_bf16(vr2.x, vr2.y);
      vl[(lc0 + 8) * 72] = (unsigned short)pv;  vl[(lc0 + 9) * 72] = (unsigned short)(pv >> 16);
      pv = pk_bf16(vr2.z, vr2.w);
      vl[(lc0 + 10) * 72] = (unsigned short)pv; vl[(lc0 + 11) * 72] = (unsigned short)(pv >> 16);
      pv = pk_bf16(vr3.x, vr3.y);
      vl[(lc0 + 12) * 72] = (unsigned short)pv; vl[(lc0 + 13) * 72] = (unsigned short)(pv >> 16);
      pv = pk_bf16(vr3.z, vr3.w);
      vl[(lc0 + 14) * 72] = (unsigned short)pv; vl[(lc0 + 15) * 72] = (unsigned short)(pv >> 16);
    }
    __syncthreads();

    if (kt + 1 < NT) {  // issue next tile's loads; consumed next iteration
      const float* kn = kl0 + (kt + 1) * tileStride;
      const float* vn = vl0 + (kt + 1) * tileStride;
      kr0 = *(const float4*)(kn + 0);  kr1 = *(const float4*)(kn + 4);
      kr2 = *(const float4*)(kn + 8);  kr3 = *(const float4*)(kn + 12);
      vr0 = *(const float4*)(vn + 0);  vr1 = *(const float4*)(vn + 4);
      vr2 = *(const float4*)(vn + 8);  vr3 = *(const float4*)(vn + 12);
    }

    // S^T = K * Q^T: 4 key-tiles of 16; lane(hi,m): key=16t+j+8hi, qrow=m
    float p[4][8];
#pragma unroll
    for (int t = 0; t < 4; ++t) {
      v8f acc;
#pragma unroll
      for (int j = 0; j < 8; ++j) acc[j] = 0.0f;
#pragma unroll
      for (int f = 0; f < 2; ++f) {
        ABFrag kf;  // A-frag: row = key 16t+m, K-dim = d block f
        const int kbase = (t * 16 + m) * KP + f * 16 + 8 * hi;
        kf.q[0] = *(const uint4*)&Kd[kbase];
        kf.q[1] = *(const uint4*)&Kd[kbase + 4];
        acc = __builtin_amdgcn_wmma_f32_16x16x32_bf16(
            false, kf.v, false, qf[f].v, (short)0, acc, false, false);
      }
#pragma unroll
      for (int j = 0; j < 8; ++j) p[t][j] = acc[j];
    }

    // online softmax in base-2; one scalar of state per lane (query row m)
    float rm = p[0][0];
#pragma unroll
    for (int t = 0; t < 4; ++t)
#pragma unroll
      for (int j = 0; j < 8; ++j) rm = fmaxf(rm, p[t][j]);
    rm = fmaxf(rm, __shfl_xor(rm, 16, 32));

    const float mn = fmaxf(rowmax, rm);
    const float alpha = fast_exp2(rowmax - mn);
    rowmax = mn;

    float ls = 0.0f;
#pragma unroll
    for (int t = 0; t < 4; ++t)
#pragma unroll
      for (int j = 0; j < 8; ++j) { p[t][j] = fast_exp2(p[t][j] - mn); ls += p[t][j]; }
    ls += __shfl_xor(ls, 16, 32);
    rowsum = rowsum * alpha + ls;

#pragma unroll
    for (int cb = 0; cb < 4; ++cb)
#pragma unroll
      for (int j = 0; j < 8; ++j) o[cb][j] *= alpha;

    // O^T += V^T * P^T over two 32-key blocks
#pragma unroll
    for (int kb2 = 0; kb2 < 2; ++kb2) {
      ABFrag pf;  // B-frag built in registers: one half-wave exchange
#pragma unroll
      for (int q = 0; q < 4; ++q) {
        const unsigned pe = pk_bf16(p[2 * kb2][2 * q],     p[2 * kb2][2 * q + 1]);
        const unsigned po = pk_bf16(p[2 * kb2 + 1][2 * q], p[2 * kb2 + 1][2 * q + 1]);
        const unsigned send = hi ? pe : po;
        const unsigned recv = __shfl_xor(send, 16, 32);
        pf.u[q]     = hi ? recv : pe;
        pf.u[q + 4] = hi ? po : recv;
      }
#pragma unroll
      for (int cb = 0; cb < 4; ++cb) {
        ABFrag vf;  // A-frag: row = channel cb*16+m, K-dim = keys kb2*32..+31
        const int vbase = (cb * 16 + m) * VP + kb2 * 16 + 8 * hi;
        vf.q[0] = *(const uint4*)&Vd[vbase];
        vf.q[1] = *(const uint4*)&Vd[vbase + 4];
        o[cb] = __builtin_amdgcn_wmma_f32_16x16x32_bf16(
            false, vf.v, false, pf.v, (short)0, o[cb], false, false);
      }
    }
  }

  // finalize: per-lane scalar division, vectorized store, channel sums
  const float invs = 1.0f / rowsum;
  const long rowbase = (((long)(bb * nseg + seg) * SD + qrow0 + m) * G + hg) * D;
#pragma unroll
  for (int cb = 0; cb < 4; ++cb) {
    float val[8];
#pragma unroll
    for (int j = 0; j < 8; ++j) val[j] = o[cb][j] * invs;
    *(float4*)&og[rowbase + cb * 16 + 8 * hi] =
        make_float4(val[0], val[1], val[2], val[3]);
    *(float4*)&og[rowbase + cb * 16 + 8 * hi + 4] =
        make_float4(val[4], val[5], val[6], val[7]);
#pragma unroll
    for (int d = 1; d < 16; d <<= 1)
#pragma unroll
      for (int j = 0; j < 8; ++j) val[j] += __shfl_xor(val[j], d, 32);
    if (m == 0) {
#pragma unroll
      for (int j = 0; j < 8; ++j)
        atomicAdd(&sums[(bb * G + hg) * D + cb * 16 + 8 * hi + j], val[j]);
    }
  }
}

// ---------------------------------------------------------------------------
// Single-pass finalize: every output element written exactly once.
// ---------------------------------------------------------------------------
__global__ __launch_bounds__(256) void dilfa_finalize_27685(
    const float* __restrict__ ogall, const char* __restrict__ wsbase,
    float* __restrict__ out, int total4)
{
  int tid = blockIdx.x * 256 + threadIdx.x;
  if (tid >= total4) return;
  int t = tid;
  const int c4 = t & 15;   t >>= 4;
  const int h  = t % 12;   t /= 12;
  const int n  = t & 8191; t >>= 13;
  const int bb = t;
  const int g  = h >> 2, hg = h & 3;
  const int logs = 11 + g;
  const int seg  = n >> logs;
  const int pos  = n & ((1 << logs) - 1);
  const int d    = pos - g;  // dilation offset for group g is g
  float4 res = make_float4(0.f, 0.f, 0.f, 0.f);
  if (d >= 0 && (d & ((1 << g) - 1)) == 0) {
    const int j = d >> g;
    const int nseg = 4 >> g;
    const long goff = (g == 0) ? 0 : (g == 1 ? 1048576 : 1572864);  // float4 units
    const float4 ov =
        ((const float4*)ogall)[goff + (((long)(bb * nseg + seg) * 2048 + j) * 4 + hg) * 16 + c4];
    const float4 sv = *((const float4*)(wsbase + g * 2048) + (bb * 4 + hg) * 16 + c4);
    res = make_float4(ov.x / (3.0f * sv.x), ov.y / (3.0f * sv.y),
                      ov.z / (3.0f * sv.z), ov.w / (3.0f * sv.w));
  }
  ((float4*)out)[((long)(bb * 8192 + n) * 12 + h) * 16 + c4] = res;
}

// ---------------------------------------------------------------------------
// Fallback path (small workspace): accumulate scatter per group.
// ---------------------------------------------------------------------------
__global__ __launch_bounds__(256) void dilfa_scatter_27685(
    const float* __restrict__ og, const float* __restrict__ sums,
    float* __restrict__ out, int s, int r, int off, int hbase, int nseg,
    int total4)
{
  int tid = blockIdx.x * 256 + threadIdx.x;
  if (tid >= total4) return;
  int t = tid;
  const int c4 = t & 15;   t >>= 4;
  const int hg = t & 3;    t >>= 2;
  const int j  = t & 2047; t >>= 11;
  const int seg = t % nseg;
  const int bb  = t / nseg;

  const float4 ov = ((const float4*)og)[(((long)(bb * nseg + seg) * 2048 + j) * 4 + hg) * 16 + c4];
  const float4 sv = ((const float4*)sums)[(bb * 4 + hg) * 16 + c4];
  const long oi = ((long)(bb * 8192 + seg * s + off + j * r) * 12 + hbase + hg) * 16 + c4;
  float4 cur = ((float4*)out)[oi];
  cur.x += ov.x / (3.0f * sv.x);
  cur.y += ov.y / (3.0f * sv.y);
  cur.z += ov.z / (3.0f * sv.z);
  cur.w += ov.w / (3.0f * sv.w);
  ((float4*)out)[oi] = cur;
}

__global__ __launch_bounds__(256) void dilfa_zero_27685(float4* p, int n4) {
  int i = blockIdx.x * 256 + threadIdx.x;
  if (i < n4) p[i] = make_float4(0.f, 0.f, 0.f, 0.f);
}

extern "C" void kernel_launch(void* const* d_in, const int* in_sizes, int n_in,
                              void* d_out, int out_size, void* d_ws, size_t ws_size,
                              hipStream_t stream) {
  const float* q = (const float*)d_in[0];
  const float* k = (const float*)d_in[1];
  const float* v = (const float*)d_in[2];
  float* out = (float*)d_out;

  char* ws = (char*)d_ws;
  const int segs[3]  = {2048, 4096, 8192};
  const int rates[3] = {1, 2, 4};
  const size_t ogBytes[3] = {16777216u, 8388608u, 4194304u};
  const size_t needA = 8192 + 16777216u + 8388608u + 4194304u;  // ~29.4 MB

  // zero the three 512-float sum buffers
  dilfa_zero_27685<<<2, 256, 0, stream>>>((float4*)ws, 1536 / 4);

  if (ws_size >= needA) {
    // Path A: all three O buffers kept; single finalize pass, no zero/RMW.
    float* ogall = (float*)(ws + 8192);
    size_t ogOff = 0;
    for (int g = 0; g < 3; ++g) {
      const int s = segs[g], r = rates[g];
      const int nseg = 8192 / s;
      float* sums = (float*)(ws + (size_t)g * 2048);
      dim3 grid(16, 4, 2 * nseg);
      dilfa_flash_27685<<<grid, 256, 0, stream>>>(
          q, k, v, (float*)((char*)ogall + ogOff), sums, s, r, g % r, g * 4, nseg);
      ogOff += ogBytes[g];
    }
    const int total4 = 2 * 8192 * 12 * 16;
    dilfa_finalize_27685<<<(total4 + 255) / 256, 256, 0, stream>>>(
        ogall, (const char*)ws, out, total4);
  } else {
    // Path B: reuse one O buffer; zero output then accumulate per group.
    float* og = (float*)(ws + 8192);
    const int n4 = out_size / 4;
    dilfa_zero_27685<<<(n4 + 255) / 256, 256, 0, stream>>>((float4*)out, n4);
    for (int g = 0; g < 3; ++g) {
      const int s = segs[g], r = rates[g];
      const int nseg = 8192 / s;
      float* sums = (float*)(ws + (size_t)g * 2048);
      dim3 grid(16, 4, 2 * nseg);
      dilfa_flash_27685<<<grid, 256, 0, stream>>>(q, k, v, og, sums,
                                                  s, r, g % r, g * 4, nseg);
      const int total4 = 2 * nseg * 2048 * 4 * 16;
      dilfa_scatter_27685<<<(total4 + 255) / 256, 256, 0, stream>>>(
          og, sums, out, s, r, g % r, g * 4, nseg, total4);
    }
  }
}
